// Head_16071767622318
// MI455X (gfx1250) — matrix operand order repaired and checked by reading, not compiled, not run on hardware
//
#include <hip/hip_runtime.h>
#include <hip/hip_bf16.h>

// CDNA5 (gfx1250) fused single-head causal attention.
//   x:[B,T,1024] f32, Wq/Wk/Wv:[64,1024] f32  ->  out:[B,T,64] f32
// Phase 0: one-shot conversion of Wq|Wk|Wv to f16 [192][1024] in ws.
// Phase 1: QKV projection via v_wmma_f32_16x16x32_f16; ONE wave per 16-row
//          M-tile computes all 12 output tiles (3 proj x 4 ntiles): the x
//          A-fragment is converted once and fed to 12 WMMAs per k-step,
//          B fragments are pure f16 loads (no cvt in the hot loop).
//          Q pre-scaled by 1/8; K row-major f16; V transposed f16 in ws.
// Phase 2: flash-attention per 16 query rows, key blocks of 32, online
//          softmax with half-wave shuffles, P relayout C-frag->A-frag via
//          LDS, global_prefetch of the next key block's K/V tiles.

typedef __attribute__((ext_vector_type(16))) _Float16 v16h;
typedef __attribute__((ext_vector_type(8)))  float    v8f;

#define T_LEN   4096
#define B_SZ    4
#define C_DIM   1024
#define H_DIM   64
#define NEG_BIG (-1e30f)

// ---- fragment index helpers (16-bit A 16x32 / B 32x16 layouts, wave32) ----
__device__ __forceinline__ int a_koff(int v, int half) {
    // A (MxK): VGPR v<4 -> k = 2v + 8*half ; v>=4 -> k = 16 + 2(v-4) + 8*half
    return (v < 4 ? 2 * v : 16 + 2 * (v - 4)) + 8 * half;
}
__device__ __forceinline__ int b_koff(int v, int half) {
    // B (KxN): VGPR v<4 -> k = 2v + 8*half ; v>=4 -> k = 16 + 2(v-4) + 8*half
    return (v < 4 ? 2 * v : 16 + 2 * (v - 4)) + 8 * half;
}

// =====================  Kernel 0: W -> f16  =====================
// wf layout: [proj*64 + h][1024], proj in {q,k,v}. 192*1024 elements.
__global__ __launch_bounds__(256)
void wcvt_kernel(const float* __restrict__ Wq,
                 const float* __restrict__ Wk,
                 const float* __restrict__ Wv,
                 _Float16* __restrict__ wf)
{
    const int i = blockIdx.x * 256 + threadIdx.x;   // 0 .. 192*1024-1
    const int proj = i >> 16;                       // 64*1024 = 65536 per proj
    const int off  = i & 0xFFFF;
    const float* W = (proj == 0) ? Wq : (proj == 1 ? Wk : Wv);
    wf[i] = (_Float16)W[off];
}

// =====================  Kernel 1: QKV projection  =====================
// grid.x = (B*T)/16 = 1024 M-tiles, block = 32 (one wave does all 12 tiles).
__global__ __launch_bounds__(32)
void qkv_wmma_kernel(const float* __restrict__ x,
                     const _Float16* __restrict__ wf,  // [192][1024] f16
                     _Float16* __restrict__ qf,   // [B*T][64] (scaled by 0.125)
                     _Float16* __restrict__ kf,   // [B*T][64]
                     _Float16* __restrict__ vt)   // [B][64][T] (transposed)
{
    const int lane = threadIdx.x;        // 0..31
    const int lid  = lane & 15;
    const int half = lane >> 4;
    const int mt   = blockIdx.x;         // M tile

    const float* xrow = x + (size_t)(mt * 16 + lid) * C_DIM;   // A row (m = lid)

    // B-operand row pointers: combo = proj*4 + nt, B col n = lid.
    const _Float16* wrow[12];
#pragma unroll
    for (int c = 0; c < 12; ++c) {
        const int proj = c >> 2;
        const int nt   = c & 3;
        wrow[c] = wf + (size_t)(proj * 64 + nt * 16 + lid) * C_DIM;
    }

    v8f acc[12];
#pragma unroll
    for (int c = 0; c < 12; ++c)
#pragma unroll
        for (int i = 0; i < 8; ++i) acc[c][i] = 0.0f;

    for (int kk = 0; kk < C_DIM / 32; ++kk) {
        const int K0 = kk * 32;

        // A fragment: convert x fp32 -> f16 (only conversions in hot loop)
        v16h a;
#pragma unroll
        for (int v = 0; v < 8; ++v) {
            const int ka = K0 + a_koff(v, half);
            a[2 * v]     = (_Float16)xrow[ka];
            a[2 * v + 1] = (_Float16)xrow[ka + 1];
        }

#pragma unroll
        for (int c = 0; c < 12; ++c) {
            v16h b;
#pragma unroll
            for (int v = 0; v < 8; ++v) {
                const int kb = K0 + b_koff(v, half);
                b[2 * v]     = wrow[c][kb];       // adjacent f16 -> b32 loads
                b[2 * v + 1] = wrow[c][kb + 1];
            }
            acc[c] = __builtin_amdgcn_wmma_f32_16x16x32_f16(false, a, false, b,
                                                            (short)0, acc[c],
                                                            false, false);
        }
    }

    // C/D layout: lane (n=lid, half) holds rows m = v + 8*half.
    const int rowbase = mt * 16;
    const int bidx  = rowbase / T_LEN;
    const int tbase = rowbase % T_LEN;
#pragma unroll
    for (int c = 0; c < 12; ++c) {
        const int proj = c >> 2;
        const int nt   = c & 3;
        const int h    = nt * 16 + lid;
#pragma unroll
        for (int v = 0; v < 8; ++v) {
            const int t = tbase + v + 8 * half;
            if (proj == 0) {
                qf[(size_t)(bidx * T_LEN + t) * H_DIM + h] =
                    (_Float16)(acc[c][v] * 0.125f);
            } else if (proj == 1) {
                kf[(size_t)(bidx * T_LEN + t) * H_DIM + h] = (_Float16)acc[c][v];
            } else {
                vt[((size_t)(bidx * H_DIM + h)) * T_LEN + t] = (_Float16)acc[c][v];
            }
        }
    }
}

// =====================  Kernel 2: flash attention  =====================
// grid.x = T/16 = 256 query tiles, grid.y = B, block = 32 (one wave).
__global__ __launch_bounds__(32)
void flash_wmma_kernel(const _Float16* __restrict__ qf,
                       const _Float16* __restrict__ kf,
                       const _Float16* __restrict__ vt,
                       float* __restrict__ out)
{
    __shared__ _Float16 pLds[16][32];    // P tile staging (C-frag -> A-frag)

    const int lane = threadIdx.x;
    const int lid  = lane & 15;
    const int half = lane >> 4;
    const int qt   = blockIdx.x;
    const int b    = blockIdx.y;
    const int qbase = qt * 16;

    // ---- load Q fragments (A operand, two 32-wide chunks of H=64) ----
    const _Float16* qrow = qf + (size_t)(b * T_LEN + qbase + lid) * H_DIM;
    v16h aq0, aq1;
#pragma unroll
    for (int v = 0; v < 8; ++v) {
        const int k = a_koff(v, half);
        aq0[2 * v]     = qrow[k];
        aq0[2 * v + 1] = qrow[k + 1];
        aq1[2 * v]     = qrow[32 + k];
        aq1[2 * v + 1] = qrow[32 + k + 1];
    }

    float mrow[8], lrow[8];
    v8f acc[4];
#pragma unroll
    for (int v = 0; v < 8; ++v) { mrow[v] = NEG_BIG; lrow[v] = 0.0f; }
#pragma unroll
    for (int nt = 0; nt < 4; ++nt)
#pragma unroll
        for (int v = 0; v < 8; ++v) acc[nt][v] = 0.0f;

    const int nkb = (qt + 2) >> 1;       // ceil((qbase+16)/32) causal key blocks

    for (int kb = 0; kb < nkb; ++kb) {
        const int kb0 = kb * 32;

        // ---- prefetch next key block's K and V tiles into cache ----
        if (kb + 1 < nkb) {
            const int nb0 = kb0 + 32;
            // 32 K rows (128 B each): one row per lane
            __builtin_prefetch(kf + (size_t)(b * T_LEN + nb0 + lane) * H_DIM, 0, 3);
            // 64 Vt row segments (64 B each): two rows per lane
            __builtin_prefetch(vt + ((size_t)(b * H_DIM + lane)) * T_LEN + nb0, 0, 3);
            __builtin_prefetch(vt + ((size_t)(b * H_DIM + 32 + lane)) * T_LEN + nb0, 0, 3);
        }

        // ---- S = Q * K^T : two 16x16 tiles (keys kb0..+15, kb0+16..+31) ----
        v8f s1, s2;
#pragma unroll
        for (int i = 0; i < 8; ++i) { s1[i] = 0.0f; s2[i] = 0.0f; }

#pragma unroll
        for (int s = 0; s < 2; ++s) {
            const _Float16* krow =
                kf + (size_t)(b * T_LEN + kb0 + 16 * s + lid) * H_DIM;
            v16h bk0, bk1;
#pragma unroll
            for (int v = 0; v < 8; ++v) {
                const int k = b_koff(v, half);
                bk0[2 * v]     = krow[k];
                bk0[2 * v + 1] = krow[k + 1];
                bk1[2 * v]     = krow[32 + k];
                bk1[2 * v + 1] = krow[32 + k + 1];
            }
            if (s == 0) {
                s1 = __builtin_amdgcn_wmma_f32_16x16x32_f16(false, aq0, false, bk0,
                                                            (short)0, s1, false, false);
                s1 = __builtin_amdgcn_wmma_f32_16x16x32_f16(false, aq1, false, bk1,
                                                            (short)0, s1, false, false);
            } else {
                s2 = __builtin_amdgcn_wmma_f32_16x16x32_f16(false, aq0, false, bk0,
                                                            (short)0, s2, false, false);
                s2 = __builtin_amdgcn_wmma_f32_16x16x32_f16(false, aq1, false, bk1,
                                                            (short)0, s2, false, false);
            }
        }

        // ---- causal mask + online softmax update ----
        const int key1 = kb0 + lid;
        const int key2 = kb0 + 16 + lid;
#pragma unroll
        for (int v = 0; v < 8; ++v) {
            const int qrow_g = qbase + v + 8 * half;
            if (key1 > qrow_g) s1[v] = NEG_BIG;
            if (key2 > qrow_g) s2[v] = NEG_BIG;
        }

#pragma unroll
        for (int v = 0; v < 8; ++v) {
            float rmax = fmaxf(s1[v], s2[v]);
#pragma unroll
            for (int off = 1; off < 16; off <<= 1)
                rmax = fmaxf(rmax, __shfl_xor(rmax, off, 32));   // half-wave reduce
            const float mnew  = fmaxf(mrow[v], rmax);
            const float alpha = __expf(mrow[v] - mnew);
            mrow[v] = mnew;
            const float p1 = __expf(s1[v] - mnew);
            const float p2 = __expf(s2[v] - mnew);
            s1[v] = p1; s2[v] = p2;
            float rsum = p1 + p2;
#pragma unroll
            for (int off = 1; off < 16; off <<= 1)
                rsum += __shfl_xor(rsum, off, 32);
            lrow[v] = lrow[v] * alpha + rsum;
#pragma unroll
            for (int nt = 0; nt < 4; ++nt) acc[nt][v] *= alpha;
        }

        // ---- relayout P (C-frag) -> A-frag through LDS ----
        __syncthreads();
#pragma unroll
        for (int v = 0; v < 8; ++v) {
            pLds[v + 8 * half][lid]      = (_Float16)s1[v];
            pLds[v + 8 * half][16 + lid] = (_Float16)s2[v];
        }
        __syncthreads();

        v16h ap;
#pragma unroll
        for (int v = 0; v < 8; ++v) {
            const int k = a_koff(v, half);
            ap[2 * v]     = pLds[lid][k];
            ap[2 * v + 1] = pLds[lid][k + 1];
        }

        // ---- acc += P (16x32) * V_block (32x64), V stored transposed ----
#pragma unroll
        for (int nt = 0; nt < 4; ++nt) {
            const _Float16* vrow =
                vt + ((size_t)(b * H_DIM + nt * 16 + lid)) * T_LEN + kb0;
            v16h bv;
#pragma unroll
            for (int v = 0; v < 8; ++v) {
                const int k = b_koff(v, half);
                bv[2 * v]     = vrow[k];
                bv[2 * v + 1] = vrow[k + 1];
            }
            acc[nt] = __builtin_amdgcn_wmma_f32_16x16x32_f16(false, ap, false, bv,
                                                             (short)0, acc[nt],
                                                             false, false);
        }
    }

    // ---- normalize (one rcp per row) and store fp32 output ----
#pragma unroll
    for (int v = 0; v < 8; ++v) {
        const float inv = 1.0f / lrow[v];
        const int t = qbase + v + 8 * half;
#pragma unroll
        for (int nt = 0; nt < 4; ++nt) {
            const int h = nt * 16 + lid;
            out[(size_t)(b * T_LEN + t) * H_DIM + h] = acc[nt][v] * inv;
        }
    }
}

extern "C" void kernel_launch(void* const* d_in, const int* in_sizes, int n_in,
                              void* d_out, int out_size, void* d_ws, size_t ws_size,
                              hipStream_t stream) {
    const float* x  = (const float*)d_in[0];
    const float* Wq = (const float*)d_in[1];
    const float* Wk = (const float*)d_in[2];
    const float* Wv = (const float*)d_in[3];
    float* out = (float*)d_out;

    const size_t BT = (size_t)B_SZ * T_LEN;          // 16384 rows
    _Float16* qf = (_Float16*)d_ws;                  // BT*64 f16   (2 MB)
    _Float16* kf = qf + BT * H_DIM;                  // BT*64 f16   (2 MB)
    _Float16* vt = kf + BT * H_DIM;                  // B*64*T f16  (2 MB)
    _Float16* wf = vt + BT * H_DIM;                  // 192*1024 f16 (384 KB)

    // Phase 0: convert Wq|Wk|Wv to f16 (192*1024 elems, 768 blocks x 256)
    wcvt_kernel<<<dim3((3 * H_DIM * C_DIM) / 256, 1, 1), dim3(256, 1, 1), 0,
                  stream>>>(Wq, Wk, Wv, wf);

    // Phase 1: QKV projection (1024 M-tiles, 1 wave computes 12 tiles each)
    qkv_wmma_kernel<<<dim3(BT / 16, 1, 1), dim3(32, 1, 1), 0, stream>>>(
        x, wf, qf, kf, vt);

    // Phase 2: flash attention (256 query tiles x 4 batches, 1 wave each)
    flash_wmma_kernel<<<dim3(T_LEN / 16, B_SZ, 1), dim3(32, 1, 1), 0, stream>>>(
        qf, kf, vt, out);
}
